// Model_59760174957273
// MI455X (gfx1250) — compile-verified, matrix operand order checked by reading, not run on hardware
//
#include <hip/hip_runtime.h>
#include <stdint.h>

#define N_NODES 100000
#define N_DST   50000
#define N_REL   32
#define DMODEL  128
#define NHEAD   4
#define NEDGE   3200000
#define DH      (DMODEL * NHEAD)   // 512

// All GEMM M dimensions are multiples of 16 -> no tail tiles, no store guards.
static_assert(N_NODES % 16 == 0 && N_DST % 16 == 0, "full m-tiles assumed");

typedef __attribute__((ext_vector_type(16))) __bf16    v16bf;
typedef __attribute__((ext_vector_type(8)))  float     v8f;
typedef __attribute__((ext_vector_type(4)))  uint32_t  u32x4;

// ---------- bf16 helpers (round-to-nearest-even) ----------
__device__ __forceinline__ uint16_t f2bf(float f) {
    uint32_t u = __float_as_uint(f);
    uint32_t r = u + 0x7FFFu + ((u >> 16) & 1u);
    return (uint16_t)(r >> 16);
}
__device__ __forceinline__ float bf2f(uint16_t b) {
    return __uint_as_float(((uint32_t)b) << 16);
}
// orderable uint encoding of float for atomicMax-based segment max
__device__ __forceinline__ uint32_t enc_f32(float f) {
    uint32_t u = __float_as_uint(f);
    return (u & 0x80000000u) ? ~u : (u | 0x80000000u);
}
__device__ __forceinline__ float dec_f32(uint32_t k) {
    uint32_t u = (k & 0x80000000u) ? (k ^ 0x80000000u) : ~k;
    return __uint_as_float(u);
}
#define ENC_NEGINF 0x007FFFFFu   // enc(-inf)

// ---------- init: zero g output, init segment max/sum ----------
__global__ void k_init(float* __restrict__ g_out, uint32_t* __restrict__ menc,
                       float* __restrict__ ssum) {
    int idx = blockIdx.x * blockDim.x + threadIdx.x;
    if (idx < N_DST * DMODEL) g_out[idx] = 0.0f;
    if (idx < N_DST * NHEAD) { menc[idx] = ENC_NEGINF; ssum[idx] = 0.0f; }
}

// ---------- gather x = node_emb[src_ids]; bf16 for GEMM, f32 rows[:N_DST] ----------
__global__ void k_gather(const int* __restrict__ src_ids,
                         const float* __restrict__ node_emb,
                         uint16_t* __restrict__ x_bf, float* __restrict__ x_f) {
    int idx = blockIdx.x * blockDim.x + threadIdx.x;
    if (idx >= N_NODES * DMODEL) return;
    int i = idx >> 7, d = idx & 127;
    float v = node_emb[src_ids[i] * DMODEL + d];
    x_bf[idx] = f2bf(v);
    if (i < N_DST) x_f[idx] = v;
}

// ---------- transpose + convert weights: w(K,N) row-major -> wT(N,K) bf16 ----------
__global__ void k_wT(const float* __restrict__ w, uint16_t* __restrict__ wT, int K, int N) {
    int idx = blockIdx.x * blockDim.x + threadIdx.x;
    if (idx >= K * N) return;
    int k = idx / N, n = idx % N;
    wT[n * K + k] = f2bf(w[idx]);
}

// ---------- WMMA fragment union ----------
union Frag { u32x4 q[2]; v16bf v; };

// ISA layout (CDNA5 ISA §7.12.2): lane<16 -> k elems {0..7,16..23}, lane>=16 -> +8.
// One wave owns a 16-row m-tile; full-K A fragment stays in registers (32 VGPRs)
// and is reused across ALL n-tiles -> A traffic cut by tiles_n (32x for DH).

// ---------- feat GEMM: out_bf16(M,N) = A(M,128) * B(128,N) + bias, via BT(N,128) ----------
__global__ void k_gemm_feat(const uint16_t* __restrict__ A, const uint16_t* __restrict__ BT,
                            const float* __restrict__ bias, uint16_t* __restrict__ out,
                            int M, int N) {
    int wid  = (blockIdx.x * blockDim.x + threadIdx.x) >> 5;
    int lane = threadIdx.x & 31;
    int tiles_m = M >> 4;                                    // M % 16 == 0
    if (wid >= tiles_m) return;                              // wave-uniform exit
    int m0 = wid << 4;
    int row = m0 + (lane & 15);
    int koff = (lane >> 4) << 3;                             // 0 or 8
    const uint16_t* arow = A + (size_t)row * DMODEL;

    Frag a[4];
#pragma unroll
    for (int t = 0; t < 4; ++t) {
        a[t].q[0] = *(const u32x4*)(arow + 32 * t + koff);
        a[t].q[1] = *(const u32x4*)(arow + 32 * t + 16 + koff);
    }

    int tiles_n = N >> 4;
    int rbase = m0 + ((lane >> 4) << 3);
    for (int nt = 0; nt < tiles_n; ++nt) {
        int col = (nt << 4) + (lane & 15);
        const uint16_t* brow = BT + (size_t)col * DMODEL;
        v8f acc;
#pragma unroll
        for (int r = 0; r < 8; ++r) acc[r] = 0.0f;
#pragma unroll
        for (int t = 0; t < 4; ++t) {
            Frag b;
            b.q[0] = *(const u32x4*)(brow + 32 * t + koff);
            b.q[1] = *(const u32x4*)(brow + 32 * t + 16 + koff);
            acc = __builtin_amdgcn_wmma_f32_16x16x32_bf16(
                false, a[t].v, false, b.v, (short)0, acc, false, false);
        }
        float bv = bias[col];
        uint16_t* op = out + (size_t)rbase * N + col;
#pragma unroll
        for (int r = 0; r < 8; ++r)                          // full tile: unguarded stores
            op[(size_t)r * N] = f2bf(acc[r] + bv);
    }
}

// ---------- final GEMM: leaky(0.01)(A*B + bias) to two f32 outputs ----------
__global__ void k_gemm_final(const uint16_t* __restrict__ A, const uint16_t* __restrict__ BT,
                             const float* __restrict__ bias,
                             float* __restrict__ out1, float* __restrict__ out2,
                             int M, int N) {
    int wid  = (blockIdx.x * blockDim.x + threadIdx.x) >> 5;
    int lane = threadIdx.x & 31;
    int tiles_m = M >> 4;                                    // M % 16 == 0
    if (wid >= tiles_m) return;
    int m0 = wid << 4;
    int row = m0 + (lane & 15);
    int koff = (lane >> 4) << 3;
    const uint16_t* arow = A + (size_t)row * DMODEL;

    Frag a[4];
#pragma unroll
    for (int t = 0; t < 4; ++t) {
        a[t].q[0] = *(const u32x4*)(arow + 32 * t + koff);
        a[t].q[1] = *(const u32x4*)(arow + 32 * t + 16 + koff);
    }

    int tiles_n = N >> 4;
    int rbase = m0 + ((lane >> 4) << 3);
    for (int nt = 0; nt < tiles_n; ++nt) {
        int col = (nt << 4) + (lane & 15);
        const uint16_t* brow = BT + (size_t)col * DMODEL;
        v8f acc;
#pragma unroll
        for (int r = 0; r < 8; ++r) acc[r] = 0.0f;
#pragma unroll
        for (int t = 0; t < 4; ++t) {
            Frag b;
            b.q[0] = *(const u32x4*)(brow + 32 * t + koff);
            b.q[1] = *(const u32x4*)(brow + 32 * t + 16 + koff);
            acc = __builtin_amdgcn_wmma_f32_16x16x32_bf16(
                false, a[t].v, false, b.v, (short)0, acc, false, false);
        }
        float bv = bias[col];
        size_t base = (size_t)rbase * N + col;
#pragma unroll
        for (int r = 0; r < 8; ++r) {
            float v = acc[r] + bv;
            v = (v >= 0.0f) ? v : 0.01f * v;
            out1[base + (size_t)r * N] = v;
            out2[base + (size_t)r * N] = v;
        }
    }
}

// ---------- edge logits: grid-stride, one wave per edge, prefetch next edge's rows ----------
__global__ void k_edge_logits(const int* __restrict__ edge_src, const int* __restrict__ edge_dst,
                              const uint16_t* __restrict__ feat_src,
                              const uint16_t* __restrict__ feat_dst,
                              const float* __restrict__ attn,
                              float* __restrict__ logits, uint32_t* __restrict__ menc) {
    int tid  = blockIdx.x * blockDim.x + threadIdx.x;
    int lane = tid & 31;
    int wave = tid >> 5;
    int nwaves = (int)((gridDim.x * (size_t)blockDim.x) >> 5);
    for (int e = wave; e < NEDGE; e += nwaves) {
        int e2 = e + nwaves;                                  // prefetch next iteration's rows:
        if (e2 < NEDGE) {                                     // 32 lanes x 32B = full 1KB row
            const char* pf = (const char*)(feat_src + (size_t)edge_src[e2] * DH) + lane * 32;
            __builtin_prefetch(pf, 0, 1);
            const char* pd = (const char*)(feat_dst + (size_t)edge_dst[e2] * DH) + lane * 32;
            __builtin_prefetch(pd, 0, 1);
        }
        int src = edge_src[e], dst = edge_dst[e];
        const uint16_t* fs = feat_src + (size_t)src * DH;
        const uint16_t* fd = feat_dst + (size_t)dst * DH;
#pragma unroll
        for (int h = 0; h < NHEAD; ++h) {
            float acc = 0.0f;
#pragma unroll
            for (int i = 0; i < 4; ++i) {
                int d = lane + 32 * i;
                float v = bf2f(fs[h * DMODEL + d]) + bf2f(fd[h * DMODEL + d]);
                v = (v >= 0.0f) ? v : 0.2f * v;               // NEG_SLOPE
                acc += v * attn[h * DMODEL + d];
            }
#pragma unroll
            for (int m = 16; m > 0; m >>= 1) acc += __shfl_xor(acc, m, 32);
            if (lane == 0) {
                logits[e * NHEAD + h] = acc;
                atomicMax(&menc[dst * NHEAD + h], enc_f32(acc));
            }
        }
    }
}

// ---------- exp(logit - m) and segment sum ----------
__global__ void k_edge_exp(const int* __restrict__ edge_dst, const uint32_t* __restrict__ menc,
                           float* __restrict__ logits_ex, float* __restrict__ ssum) {
    int idx = blockIdx.x * blockDim.x + threadIdx.x;
    if (idx >= NEDGE * NHEAD) return;
    int e = idx >> 2, h = idx & 3;
    int dst = edge_dst[e];
    float m = dec_f32(menc[dst * NHEAD + h]);
    if (m < -3.0e38f) m = 0.0f;                               // non-finite segment max -> 0
    float ex = __expf(logits_ex[idx] - m);
    logits_ex[idx] = ex;
    atomicAdd(&ssum[dst * NHEAD + h], ex);
}

// ---------- a = ex / (sum + eps)  (also the `attentions` output) ----------
__global__ void k_edge_a(const int* __restrict__ edge_dst, const float* __restrict__ ex,
                         const float* __restrict__ ssum, float* __restrict__ a_out) {
    int idx = blockIdx.x * blockDim.x + threadIdx.x;
    if (idx >= NEDGE * NHEAD) return;
    int e = idx >> 2, h = idx & 3;
    int dst = edge_dst[e];
    a_out[idx] = ex[idx] / (ssum[dst * NHEAD + h] + 1e-16f);
}

// ---------- message: softmax_D(en*rr*a), sum heads, atomic-accumulate into g ----------
__global__ void k_message(const int* __restrict__ edge_src, const int* __restrict__ edge_dst,
                          const int* __restrict__ edge_rtype,
                          const uint16_t* __restrict__ feat_src,
                          const float* __restrict__ rel_emb,
                          const float* __restrict__ a_vals, float* __restrict__ g_out) {
    int tid  = blockIdx.x * blockDim.x + threadIdx.x;
    int lane = tid & 31;
    int wave = tid >> 5;
    int nwaves = (int)((gridDim.x * (size_t)blockDim.x) >> 5);
    for (int e = wave; e < NEDGE; e += nwaves) {
        int e2 = e + nwaves;
        if (e2 < NEDGE) {
            const char* pf = (const char*)(feat_src + (size_t)edge_src[e2] * DH) + lane * 32;
            __builtin_prefetch(pf, 0, 1);
        }
        int src = edge_src[e], dst = edge_dst[e], rt = edge_rtype[e];
        const uint16_t* fs = feat_src + (size_t)src * DH;
        const float* rr = rel_emb + rt * DMODEL;
        float gacc[4] = {0.0f, 0.0f, 0.0f, 0.0f};
#pragma unroll
        for (int h = 0; h < NHEAD; ++h) {
            float a = a_vals[e * NHEAD + h];
            float v[4];
            float mx = -3.4e38f;
#pragma unroll
            for (int i = 0; i < 4; ++i) {
                int d = lane + 32 * i;
                v[i] = bf2f(fs[h * DMODEL + d]) * rr[d] * a;
                mx = fmaxf(mx, v[i]);
            }
#pragma unroll
            for (int m = 16; m > 0; m >>= 1) mx = fmaxf(mx, __shfl_xor(mx, m, 32));
            float sum = 0.0f;
#pragma unroll
            for (int i = 0; i < 4; ++i) { v[i] = __expf(v[i] - mx); sum += v[i]; }
#pragma unroll
            for (int m = 16; m > 0; m >>= 1) sum += __shfl_xor(sum, m, 32);
            float inv = 1.0f / sum;
#pragma unroll
            for (int i = 0; i < 4; ++i) gacc[i] += v[i] * inv;
        }
#pragma unroll
        for (int i = 0; i < 4; ++i)
            atomicAdd(&g_out[(size_t)dst * DMODEL + lane + 32 * i], gacc[i]);
    }
}

// ---------- t = bf16(h_dst + g) for the final GEMM ----------
__global__ void k_make_t(const float* __restrict__ x_f, const float* __restrict__ g,
                         uint16_t* __restrict__ t_bf) {
    int idx = blockIdx.x * blockDim.x + threadIdx.x;
    if (idx >= N_DST * DMODEL) return;
    t_bf[idx] = f2bf(x_f[idx] + g[idx]);
}

extern "C" void kernel_launch(void* const* d_in, const int* in_sizes, int n_in,
                              void* d_out, int out_size, void* d_ws, size_t ws_size,
                              hipStream_t stream) {
    const int*   src_ids    = (const int*)d_in[0];
    const int*   edge_src   = (const int*)d_in[1];
    const int*   edge_dst   = (const int*)d_in[2];
    const int*   edge_rtype = (const int*)d_in[3];
    const float* node_emb   = (const float*)d_in[4];
    const float* rel_emb    = (const float*)d_in[5];
    const float* w1_w       = (const float*)d_in[6];
    const float* w1_b       = (const float*)d_in[7];
    const float* w2s_w      = (const float*)d_in[8];
    const float* w2s_b      = (const float*)d_in[9];
    const float* w2d_w      = (const float*)d_in[10];
    const float* w2d_b      = (const float*)d_in[11];
    const float* attn       = (const float*)d_in[12];

    // workspace carve-out (256B aligned)
    char* ws = (char*)d_ws;
    size_t off = 0;
    auto carve = [&](size_t bytes) {
        char* p = ws + off;
        off = (off + bytes + 255) & ~(size_t)255;
        return p;
    };
    uint16_t* x_bf   = (uint16_t*)carve((size_t)N_NODES * DMODEL * 2);
    float*    x_f    = (float*)   carve((size_t)N_DST   * DMODEL * 4);
    uint16_t* w2sT   = (uint16_t*)carve((size_t)DMODEL * DH * 2);
    uint16_t* w2dT   = (uint16_t*)carve((size_t)DMODEL * DH * 2);
    uint16_t* w1T    = (uint16_t*)carve((size_t)DMODEL * DMODEL * 2);
    uint16_t* fsrc   = (uint16_t*)carve((size_t)N_NODES * DH * 2);
    uint16_t* fdst   = (uint16_t*)carve((size_t)N_DST   * DH * 2);
    float*    logits = (float*)   carve((size_t)NEDGE * NHEAD * 4);
    uint32_t* menc   = (uint32_t*)carve((size_t)N_DST * NHEAD * 4);
    float*    ssum   = (float*)   carve((size_t)N_DST * NHEAD * 4);
    uint16_t* t_bf   = (uint16_t*)carve((size_t)N_DST * DMODEL * 2);

    float* out_xout = (float*)d_out;
    float* out_emb  = out_xout + (size_t)N_DST * DMODEL;
    float* out_g    = out_emb  + (size_t)N_DST * DMODEL;
    float* out_attn = out_g    + (size_t)N_DST * DMODEL;

    const int T = 256;
    k_init  <<<(N_DST * DMODEL + T - 1) / T, T, 0, stream>>>(out_g, menc, ssum);
    k_gather<<<(N_NODES * DMODEL + T - 1) / T, T, 0, stream>>>(src_ids, node_emb, x_bf, x_f);
    k_wT    <<<(DMODEL * DH + T - 1) / T, T, 0, stream>>>(w2s_w, w2sT, DMODEL, DH);
    k_wT    <<<(DMODEL * DH + T - 1) / T, T, 0, stream>>>(w2d_w, w2dT, DMODEL, DH);
    k_wT    <<<(DMODEL * DMODEL + T - 1) / T, T, 0, stream>>>(w1_w, w1T, DMODEL, DMODEL);

    {   // feat_src = x @ w2s + b  (100000 x 512); one wave per 16-row m-tile
        int waves = N_NODES / 16;
        k_gemm_feat<<<(waves * 32 + T - 1) / T, T, 0, stream>>>(x_bf, w2sT, w2s_b, fsrc, N_NODES, DH);
    }
    {   // feat_dst = x[:N_DST] @ w2d + b  (50000 x 512)
        int waves = N_DST / 16;
        k_gemm_feat<<<(waves * 32 + T - 1) / T, T, 0, stream>>>(x_bf, w2dT, w2d_b, fdst, N_DST, DH);
    }

    const int EDGE_BLOCKS = 20000;   // 160000 waves, grid-stride over 3.2M edges
    k_edge_logits<<<EDGE_BLOCKS, T, 0, stream>>>(edge_src, edge_dst, fsrc, fdst, attn,
                                                 logits, menc);
    k_edge_exp<<<(NEDGE * NHEAD + T - 1) / T, T, 0, stream>>>(edge_dst, menc, logits, ssum);
    k_edge_a  <<<(NEDGE * NHEAD + T - 1) / T, T, 0, stream>>>(edge_dst, logits, ssum, out_attn);
    k_message <<<EDGE_BLOCKS, T, 0, stream>>>(edge_src, edge_dst, edge_rtype, fsrc, rel_emb,
                                              out_attn, out_g);

    k_make_t<<<(N_DST * DMODEL + T - 1) / T, T, 0, stream>>>(x_f, out_g, t_bf);
    {   // x_out = leaky0.01((h_dst+g) @ w1 + b), duplicated to emb_cat
        int waves = N_DST / 16;
        k_gemm_final<<<(waves * 32 + T - 1) / T, T, 0, stream>>>(t_bf, w1T, w1_b,
                                                                 out_xout, out_emb, N_DST, DMODEL);
    }
}
